// LlamaTopKAttention_64424509440378
// MI455X (gfx1250) — matrix operand order repaired and checked by reading, not compiled, not run on hardware
//
#include <hip/hip_runtime.h>

// ---------------------------------------------------------------------------
// LlamaTopKAttention for MI455X (gfx1250, wave32, WMMA).
// Pipeline: QKV GEMMs (f16 WMMA, f32 acc, double-buffered LDS) -> RoPE ->
// flash causal GQA attention (32 q-rows/wave, v_wmma_f32_16x16x32_f16 for
// QK^T and PV) -> Wo GEMM.  Workspace requirement: 64 MB.
// ---------------------------------------------------------------------------

typedef _Float16 h16;
typedef __attribute__((ext_vector_type(16))) _Float16 v16h;
typedef __attribute__((ext_vector_type(8)))  _Float16 v8h;
typedef __attribute__((ext_vector_type(8)))  float    v8f;

union FragA { v16h v; v8h h8[2]; };

__device__ __forceinline__ v8f vzero8f() {
    v8f z;
#pragma unroll
    for (int i = 0; i < 8; ++i) z[i] = 0.0f;
    return z;
}

#define WMMA_F16(A, B, C) \
    __builtin_amdgcn_wmma_f32_16x16x32_f16(false, (A), false, (B), (short)0, (C), false, false)

// ---------------------------------------------------------------------------
// Tiled GEMM: C(MxN) = A(MxK) * B(KxN).  A is f32 or f16, B is f32 (converted
// to f16 while staging into LDS).  MODE 0: f32 row-major store.
// MODE 1: f16 transposed store C[n*M + m] (used to produce V^T).
// Block: 256 threads (8 waves), tile 64x256, K-step 32, wave tile 32x64,
// double-buffered LDS (8KB A + 32KB B of the 320KB WGP pool).
// ---------------------------------------------------------------------------
template <int MODE, bool A_HALF>
__global__ __launch_bounds__(256)
void gemm_wmma_kernel(const void* __restrict__ Av, const float* __restrict__ B,
                      void* __restrict__ Cv, int M, int N, int K)
{
    constexpr int BM = 64, BN = 256, BK = 32;
    __shared__ __align__(16) h16 As[2][BM][BK];     // A tiles, row-major
    __shared__ __align__(16) h16 Bts[2][BN][BK];    // B tiles, transposed

    const int tid  = threadIdx.x;
    const int lane = tid & 31;
    const int wid  = tid >> 5;
    const int l16  = lane & 15;
    const int hi   = lane >> 4;
    const int wm   = (wid >> 2) << 5;               // wave M offset: 0 / 32
    const int wn   = (wid & 3) << 6;                // wave N offset: 0..192
    const int m0   = blockIdx.y * BM;
    const int n0   = blockIdx.x * BN;

    const float* Af = (const float*)Av;
    const h16*   Ah = (const h16*)Av;

    v8f acc[2][4];
#pragma unroll
    for (int i = 0; i < 2; ++i)
#pragma unroll
        for (int j = 0; j < 4; ++j) acc[i][j] = vzero8f();

    const int arow = tid >> 2;              // 0..63
    const int acol = (tid & 3) * 8;         // 0,8,16,24
    const int brow = tid >> 3;              // 0..31
    const int bcol = (tid & 7) * 32;        // 0..224

    auto stage = [&](int buf, int kk) {
        // ---- A tile (convert f32 -> f16 unless already f16) ----
        if (A_HALF) {
            v8h a = *(const v8h*)(Ah + (size_t)(m0 + arow) * K + kk + acol);
            *(v8h*)&As[buf][arow][acol] = a;
        } else {
            const float* p = Af + (size_t)(m0 + arow) * K + kk + acol;
            float4 f0 = *(const float4*)p;
            float4 f1 = *(const float4*)(p + 4);
            v8h a;
            a[0] = (h16)f0.x; a[1] = (h16)f0.y; a[2] = (h16)f0.z; a[3] = (h16)f0.w;
            a[4] = (h16)f1.x; a[5] = (h16)f1.y; a[6] = (h16)f1.z; a[7] = (h16)f1.w;
            *(v8h*)&As[buf][arow][acol] = a;
        }
        // ---- B tile, transposed into LDS ----
        const float* p = B + (size_t)(kk + brow) * N + n0 + bcol;
#pragma unroll
        for (int j = 0; j < 32; j += 4) {
            float4 f = *(const float4*)(p + j);
            Bts[buf][bcol + j + 0][brow] = (h16)f.x;
            Bts[buf][bcol + j + 1][brow] = (h16)f.y;
            Bts[buf][bcol + j + 2][brow] = (h16)f.z;
            Bts[buf][bcol + j + 3][brow] = (h16)f.w;
        }
    };

    stage(0, 0);
    __syncthreads();

    for (int k0 = 0; k0 < K; k0 += BK) {
        const int cur = (k0 >> 5) & 1;

        // overlap: stage next K tile while current one feeds the WMMAs
        if (k0 + BK < K) stage(cur ^ 1, k0 + BK);
        // warm L2 for the tile after next (global_prefetch_b8)
        if (k0 + 2 * BK < K) {
            if (A_HALF) __builtin_prefetch(Ah + (size_t)(m0 + arow) * K + k0 + 2 * BK + acol, 0, 1);
            else        __builtin_prefetch(Af + (size_t)(m0 + arow) * K + k0 + 2 * BK + acol, 0, 1);
            __builtin_prefetch(B + (size_t)(k0 + 2 * BK + brow) * N + n0 + bcol, 0, 1);
        }

        // ---- fragments (ISA 7.12.2 16-bit A layout; B via transposed LDS) ----
        FragA af[2], bf[4];
#pragma unroll
        for (int ms = 0; ms < 2; ++ms) {
            const h16* p = &As[cur][wm + ms * 16 + l16][hi * 8];
            af[ms].h8[0] = *(const v8h*)p;          // K = hi*8 .. +7
            af[ms].h8[1] = *(const v8h*)(p + 16);   // K = hi*8+16 .. +23
        }
#pragma unroll
        for (int ns = 0; ns < 4; ++ns) {
            const h16* p = &Bts[cur][wn + ns * 16 + l16][hi * 8];
            bf[ns].h8[0] = *(const v8h*)p;          // K = hi*8 .. +7
            bf[ns].h8[1] = *(const v8h*)(p + 16);   // K = hi*8+16 .. +23
        }
#pragma unroll
        for (int ms = 0; ms < 2; ++ms)
#pragma unroll
            for (int ns = 0; ns < 4; ++ns)
                acc[ms][ns] = WMMA_F16(af[ms].v, bf[ns].v, acc[ms][ns]);

        __syncthreads();
    }

    // ---- epilogue (C layout: vgpr r -> row r + 8*hi, lane&15 -> col) ----
#pragma unroll
    for (int ms = 0; ms < 2; ++ms)
#pragma unroll
        for (int ns = 0; ns < 4; ++ns)
#pragma unroll
            for (int r = 0; r < 8; ++r) {
                const int gm = m0 + wm + ms * 16 + r + hi * 8;
                const int gn = n0 + wn + ns * 16 + l16;
                const float val = acc[ms][ns][r];
                if (MODE == 0) ((float*)Cv)[(size_t)gm * N + gn] = val;
                else           ((h16*)Cv)[(size_t)gn * M + gm] = (h16)val;
            }
}

// ---------------------------------------------------------------------------
// RoPE + f32->f16 + layout change: src[s][nh*128] -> dst[h][s][128].
// Pairs (i, i+64); Q additionally folds in 1/sqrt(HD) via `scale`.
// ---------------------------------------------------------------------------
__global__ void rope_convert_kernel(const float* __restrict__ src,
                                    const int* __restrict__ pos,
                                    h16* __restrict__ dst,
                                    int nh, float scale, int S)
{
    const int idx   = blockIdx.x * blockDim.x + threadIdx.x;
    const int total = S * nh * 64;
    if (idx >= total) return;
    const int i = idx & 63;
    const int h = (idx >> 6) % nh;
    const int s = idx / (nh * 64);

    const float p  = (float)pos[s];
    // inv_freq = 10000^(-2i/128) = exp(-(2i/128)*ln(10000))
    const float inv_freq = __expf(-(float)(2 * i) * (1.0f / 128.0f) * 9.210340371976184f);
    const float ang = p * inv_freq;
    float sn, c;
    __sincosf(ang, &sn, &c);

    const float* row = src + (size_t)s * (nh * 128) + h * 128;
    const float x0 = row[i];
    const float x1 = row[i + 64];
    const float y0 = (x0 * c - x1 * sn) * scale;
    const float y1 = (x1 * c + x0 * sn) * scale;

    h16* drow = dst + ((size_t)h * S + s) * 128;
    drow[i]      = (h16)y0;
    drow[i + 64] = (h16)y1;
}

// ---------------------------------------------------------------------------
// Flash causal GQA attention.  Grid (S/128, NH), block 128 (4 waves).
// Each wave owns TWO 16-row query tiles (32 rows) so every K and V fragment
// feeds two WMMAs.  Keys processed 32 at a time: 16 QK^T + 16 PV WMMAs/block.
// Qh: [16][S][128] f16 (pre-scaled by 1/sqrt(128)), Kh: [4][S][128] f16,
// Vt: [4][128][S] f16 (transposed), AO: [S][2048] f16.
// ---------------------------------------------------------------------------
__global__ __launch_bounds__(128)
void flash_attn_kernel(const h16* __restrict__ Qh, const h16* __restrict__ Kh,
                       const h16* __restrict__ Vt, h16* __restrict__ AO)
{
    constexpr int S = 4096, HD = 128;
    const int h     = blockIdx.y;
    const int kvh   = h >> 2;
    const int wid   = threadIdx.x >> 5;
    const int lane  = threadIdx.x & 31;
    const int l16   = lane & 15;
    const int hi    = lane >> 4;
    const int qbase = blockIdx.x * 128 + wid * 32;   // multiple of 32

    __shared__ __align__(16) h16 Ps[4][2][16][32];   // per-wave P tiles (C->A relayout)

    const h16* Qp = Qh + ((size_t)h * S + qbase) * HD;
    const h16* Kp = Kh + (size_t)kvh * S * HD;
    const h16* Vp = Vt + (size_t)kvh * HD * S;

    // Q fragments: 2 q-tiles x 4 chunks of K=32, resident for the whole loop.
    FragA qf[2][4];
#pragma unroll
    for (int qi = 0; qi < 2; ++qi)
#pragma unroll
        for (int c = 0; c < 4; ++c) {
            const h16* p = Qp + (size_t)(qi * 16 + l16) * HD + c * 32 + hi * 8;
            qf[qi][c].h8[0] = *(const v8h*)p;
            qf[qi][c].h8[1] = *(const v8h*)(p + 16);
        }

    v8f o[2][8];
    float rowmax[2][8], rowsum[2][8];
#pragma unroll
    for (int qi = 0; qi < 2; ++qi) {
#pragma unroll
        for (int t = 0; t < 8; ++t) o[qi][t] = vzero8f();
#pragma unroll
        for (int r = 0; r < 8; ++r) { rowmax[qi][r] = -1e30f; rowsum[qi][r] = 0.0f; }
    }

    for (int kb = 0; kb < qbase + 32; kb += 32) {
        // ---- scores: 2 q-tiles x 2 k-tiles over HD=128 (16 WMMAs) ----
        v8f sc[2][2];
#pragma unroll
        for (int qi = 0; qi < 2; ++qi) { sc[qi][0] = vzero8f(); sc[qi][1] = vzero8f(); }
#pragma unroll
        for (int c = 0; c < 4; ++c) {
            FragA k0f, k1f;
            const h16* p0 = Kp + (size_t)(kb + l16) * HD + c * 32 + hi * 8;
            k0f.h8[0] = *(const v8h*)p0;
            k0f.h8[1] = *(const v8h*)(p0 + 16);
            const h16* p1 = Kp + (size_t)(kb + 16 + l16) * HD + c * 32 + hi * 8;
            k1f.h8[0] = *(const v8h*)p1;
            k1f.h8[1] = *(const v8h*)(p1 + 16);
#pragma unroll
            for (int qi = 0; qi < 2; ++qi) {
                sc[qi][0] = WMMA_F16(qf[qi][c].v, k0f.v, sc[qi][0]);
                sc[qi][1] = WMMA_F16(qf[qi][c].v, k1f.v, sc[qi][1]);
            }
        }

        // ---- causal mask (near-diagonal blocks only; branch is wave-uniform) ----
        if (kb + 31 > qbase) {
#pragma unroll
            for (int qi = 0; qi < 2; ++qi)
#pragma unroll
                for (int r = 0; r < 8; ++r) {
                    const int q = qbase + qi * 16 + r + hi * 8;
                    if (kb + l16 > q)      sc[qi][0][r] = -1e30f;
                    if (kb + 16 + l16 > q) sc[qi][1][r] = -1e30f;
                }
        }

        // ---- online softmax per q-tile (row reductions stay in 16-lane halves) ----
#pragma unroll
        for (int qi = 0; qi < 2; ++qi) {
            float tmax[8];
#pragma unroll
            for (int r = 0; r < 8; ++r) tmax[r] = fmaxf(sc[qi][0][r], sc[qi][1][r]);
#pragma unroll
            for (int off = 1; off <= 8; off <<= 1)
#pragma unroll
                for (int r = 0; r < 8; ++r)
                    tmax[r] = fmaxf(tmax[r], __shfl_xor(tmax[r], off, 32));

            float al[8];
#pragma unroll
            for (int r = 0; r < 8; ++r) {
                const float nm = fmaxf(rowmax[qi][r], tmax[r]);
                al[r] = __expf(rowmax[qi][r] - nm);
                rowmax[qi][r] = nm;
            }
#pragma unroll
            for (int r = 0; r < 8; ++r) {
                sc[qi][0][r] = __expf(sc[qi][0][r] - rowmax[qi][r]);
                sc[qi][1][r] = __expf(sc[qi][1][r] - rowmax[qi][r]);
            }
            float ps[8];
#pragma unroll
            for (int r = 0; r < 8; ++r) ps[r] = sc[qi][0][r] + sc[qi][1][r];
#pragma unroll
            for (int off = 1; off <= 8; off <<= 1)
#pragma unroll
                for (int r = 0; r < 8; ++r) ps[r] += __shfl_xor(ps[r], off, 32);
#pragma unroll
            for (int r = 0; r < 8; ++r) rowsum[qi][r] = rowsum[qi][r] * al[r] + ps[r];
#pragma unroll
            for (int t = 0; t < 8; ++t)
#pragma unroll
                for (int r = 0; r < 8; ++r) o[qi][t][r] *= al[r];

            // P: C-layout -> A-layout via per-wave LDS tile
#pragma unroll
            for (int r = 0; r < 8; ++r) {
                Ps[wid][qi][r + hi * 8][l16]      = (h16)sc[qi][0][r];
                Ps[wid][qi][r + hi * 8][16 + l16] = (h16)sc[qi][1][r];
            }
        }
        asm volatile("s_wait_dscnt 0" ::: "memory");   // same-wave DS store->load

        FragA pa[2];
#pragma unroll
        for (int qi = 0; qi < 2; ++qi) {
            const h16* pp = &Ps[wid][qi][l16][hi * 8];
            pa[qi].h8[0] = *(const v8h*)pp;
            pa[qi].h8[1] = *(const v8h*)(pp + 16);
        }

        // ---- PV: each V^T fragment feeds both q-tiles (16 WMMAs) ----
#pragma unroll
        for (int t = 0; t < 8; ++t) {
            FragA vb;
            const h16* vp = Vp + (size_t)(t * 16 + l16) * S + kb + hi * 8;
            vb.h8[0] = *(const v8h*)vp;
            vb.h8[1] = *(const v8h*)(vp + 16);
            o[0][t] = WMMA_F16(pa[0].v, vb.v, o[0][t]);
            o[1][t] = WMMA_F16(pa[1].v, vb.v, o[1][t]);
        }
    }

    // ---- normalize & store AO[s][h*128+d] as f16 ----
#pragma unroll
    for (int qi = 0; qi < 2; ++qi)
#pragma unroll
        for (int r = 0; r < 8; ++r) {
            const float inv = 1.0f / rowsum[qi][r];
            const int gm = qbase + qi * 16 + r + hi * 8;
            h16* orow = AO + (size_t)gm * 2048 + h * 128;
#pragma unroll
            for (int t = 0; t < 8; ++t)
                orow[t * 16 + l16] = (h16)(o[qi][t][r] * inv);
        }
}

// ---------------------------------------------------------------------------
extern "C" void kernel_launch(void* const* d_in, const int* in_sizes, int n_in,
                              void* d_out, int out_size, void* d_ws, size_t ws_size,
                              hipStream_t stream)
{
    (void)in_sizes; (void)n_in; (void)out_size; (void)ws_size;

    const float* X  = (const float*)d_in[0];   // [1,4096,2048]
    const int*   pos = (const int*)d_in[1];    // [1,4096]
    const float* Wq = (const float*)d_in[2];   // [2048,2048]
    const float* Wk = (const float*)d_in[3];   // [2048,512]
    const float* Wv = (const float*)d_in[4];   // [2048,512]
    const float* Wo = (const float*)d_in[5];   // [2048,2048]
    float* out = (float*)d_out;                // [1,4096,2048]

    const int S = 4096, HID = 2048, KV = 512;

    // Workspace layout (needs 64 MB total):
    char* ws = (char*)d_ws;
    float* Qf = (float*)(ws + 0);            // 32 MB  f32 [S][2048]  (dead after RoPE)
    float* Kf = (float*)(ws + 33554432);     //  8 MB  f32 [S][512]
    h16*   Vt = (h16*)(ws + 41943040);       //  4 MB  f16 [512][S]   (V transposed)
    h16*   Qh = (h16*)(ws + 46137344);       // 16 MB  f16 [16][S][128]
    h16*   Kh = (h16*)(ws + 62914560);       //  4 MB  f16 [4][S][128]
    h16*   AO = (h16*)(ws + 0);              // 16 MB  f16 [S][2048]  (reuses Qf region)

    const dim3 blk(256);

    // QKV projections
    gemm_wmma_kernel<0, false><<<dim3(HID / 256, S / 64), blk, 0, stream>>>(X, Wq, Qf, S, HID, HID);
    gemm_wmma_kernel<0, false><<<dim3(KV  / 256, S / 64), blk, 0, stream>>>(X, Wk, Kf, S, KV,  HID);
    gemm_wmma_kernel<1, false><<<dim3(KV  / 256, S / 64), blk, 0, stream>>>(X, Wv, Vt, S, KV,  HID);

    // RoPE + convert to f16, head-major (Q pre-scaled by 1/sqrt(128))
    const int tq = S * 16 * 64;
    rope_convert_kernel<<<(tq + 255) / 256, 256, 0, stream>>>(Qf, pos, Qh, 16, 0.08838834764831845f, S);
    const int tk = S * 4 * 64;
    rope_convert_kernel<<<(tk + 255) / 256, 256, 0, stream>>>(Kf, pos, Kh, 4, 1.0f, S);

    // Flash causal attention (32 query rows per wave)
    flash_attn_kernel<<<dim3(S / 128, 16), dim3(128), 0, stream>>>(Qh, Kh, Vt, AO);

    // Output projection
    gemm_wmma_kernel<0, true><<<dim3(HID / 256, S / 64), blk, 0, stream>>>(AO, Wo, out, S, HID, HID);
}
